// SchNetRepresentation_67654324846791
// MI455X (gfx1250) — matrix operand order repaired and checked by reading, not compiled, not run on hardware
//
#include <hip/hip_runtime.h>

#define NP      524288      // number of pairs
#define NATOMS  16384       // B*N
#define NB      128         // basis / filters
#define NRBF    20
#define CUTOFF_R 5.0f

typedef __attribute__((ext_vector_type(16))) _Float16 v16h;
typedef __attribute__((ext_vector_type(8)))  _Float16 v8h;
typedef __attribute__((ext_vector_type(8)))  float    v8f;

// -------------------------------------------------------------------------
// helpers
// -------------------------------------------------------------------------
__device__ __forceinline__ float ssp_f(float x) {
  // softplus(x) - log(2), numerically stable
  float ax = fabsf(x);
  return fmaxf(x, 0.f) + __logf(1.f + __expf(-ax)) - 0.69314718056f;
}

// load two aligned runs of 8 halves into one v16h
__device__ __forceinline__ v16h ld2x8h(const _Float16* p0, const _Float16* p1) {
  v8h a = *(const v8h*)p0;
  v8h b = *(const v8h*)p1;
  v16h r;
#pragma unroll
  for (int i = 0; i < 8; ++i) { r[i] = a[i]; r[i + 8] = b[i]; }
  return r;
}

// -------------------------------------------------------------------------
// pair geometry: d_ij -> gaussian RBF (f16, padded 20->32) + cosine cutoff
// -------------------------------------------------------------------------
__global__ void __launch_bounds__(256) pair_geom_kernel(
    const float* __restrict__ R, const int* __restrict__ idx_i,
    const int* __restrict__ idx_j, _Float16* __restrict__ rbf,
    float* __restrict__ rcut) {
  int p = blockIdx.x * 256 + threadIdx.x;
  int i = idx_i[p], j = idx_j[p];
  float dx = R[3 * i]     - R[3 * j];
  float dy = R[3 * i + 1] - R[3 * j + 1];
  float dz = R[3 * i + 2] - R[3 * j + 2];
  float d = sqrtf(dx * dx + dy * dy + dz * dz);
  const float width = CUTOFF_R / (float)(NRBF - 1);   // linspace spacing
  const float coeff = -0.5f / (width * width);
  _Float16* row = rbf + (size_t)p * 32;
#pragma unroll
  for (int k = 0; k < 32; ++k) {
    float v = 0.f;
    if (k < NRBF) { float df = d - (float)k * width; v = __expf(coeff * df * df); }
    row[k] = (_Float16)v;
  }
  float fc = 0.5f * (__cosf(d * (3.14159265358979f / CUTOFF_R)) + 1.f);
  rcut[p] = (d < CUTOFF_R) ? fc : 0.f;
}

// -------------------------------------------------------------------------
// x = emb[Z]   (writes the residual stream, which lives in d_out)
// -------------------------------------------------------------------------
__global__ void __launch_bounds__(256) embed_kernel(
    const int* __restrict__ Z, const float* __restrict__ emb,
    float* __restrict__ x) {
  int t = blockIdx.x * 256 + threadIdx.x;
  int row = t >> 7, col = t & 127;
  x[t] = emb[Z[row] * NB + col];
}

// -------------------------------------------------------------------------
// weight conversion: f32 row-major [K,N] -> f16 column-major [N][K]
// (column-major makes the WMMA B-fragment two contiguous b128 loads)
// -------------------------------------------------------------------------
__global__ void __launch_bounds__(256) convw128_kernel(
    const float* __restrict__ W, _Float16* __restrict__ Bt) {
  int t = blockIdx.x * 256 + threadIdx.x;   // 16384 threads
  int k = t >> 7, n = t & 127;
  Bt[n * 128 + k] = (_Float16)W[k * 128 + n];
}
__global__ void __launch_bounds__(256) convw1_kernel(
    const float* __restrict__ W /*[20,128]*/, _Float16* __restrict__ Bt /*[128][32]*/) {
  int t = blockIdx.x * 256 + threadIdx.x;   // 4096 threads
  int k = t & 31, n = t >> 5;
  Bt[n * 32 + k] = (k < NRBF) ? (_Float16)W[k * 128 + n] : (_Float16)0.f;
}

// -------------------------------------------------------------------------
// generic [M,128] @ [128,128] + bias WMMA GEMM, one wave per 16-row stripe.
// A fragment per ISA 16-bit A layout; B from column-major f16 weights.
// OUT_MODE: 0 = store f32, 1 = store f16, 2 = accumulate into f32 (residual)
// -------------------------------------------------------------------------
template <bool A_F16, bool DO_SSP, int OUT_MODE>
__global__ void __launch_bounds__(256) wmma_gemm_k128(
    const void* __restrict__ Ap, const _Float16* __restrict__ Bt,
    const float* __restrict__ bias, void* __restrict__ outp, int M) {
  int wave = blockIdx.x * 8 + (threadIdx.x >> 5);
  int lane = threadIdx.x & 31;
  int rowbase = wave * 16;
  if (rowbase >= M) return;

  const int col0 = lane & 15;
  const int rofs = (lane & 16) ? 8 : 0;   // C-layout row offset for upper half-wave
  const int aoff = (lane & 16) ? 8 : 0;   // A-layout K offset
  const int boff = (lane & 16) ? 8 : 0;   // B-layout K offset

  v16h afrag[4];
  if (A_F16) {
    const _Float16* A = (const _Float16*)Ap;
#pragma unroll
    for (int ks = 0; ks < 4; ++ks) {
      const _Float16* p = A + (size_t)(rowbase + col0) * NB + ks * 32 + aoff;
      afrag[ks] = ld2x8h(p, p + 16);
    }
  } else {
    const float* A = (const float*)Ap;
#pragma unroll
    for (int ks = 0; ks < 4; ++ks) {
      const float* p = A + (size_t)(rowbase + col0) * NB + ks * 32 + aoff;
      v16h a;
#pragma unroll
      for (int e = 0; e < 8; ++e) { a[e] = (_Float16)p[e]; a[e + 8] = (_Float16)p[16 + e]; }
      afrag[ks] = a;
    }
  }

#pragma unroll
  for (int nt = 0; nt < 8; ++nt) {
    int col = nt * 16 + col0;
    float bv = bias[col];
    v8f acc = {bv, bv, bv, bv, bv, bv, bv, bv};
#pragma unroll
    for (int ks = 0; ks < 4; ++ks) {
      const _Float16* bp = Bt + col * 128 + ks * 32 + boff;
      v16h b = ld2x8h(bp, bp + 16);
      acc = __builtin_amdgcn_wmma_f32_16x16x32_f16(false, afrag[ks], false, b,
                                                   (short)0, acc, false, false);
    }
#pragma unroll
    for (int r = 0; r < 8; ++r) {
      float v = acc[r];
      if (DO_SSP) v = ssp_f(v);
      size_t o = (size_t)(rowbase + r + rofs) * NB + col;
      if (OUT_MODE == 0)      ((float*)outp)[o] = v;
      else if (OUT_MODE == 1) ((_Float16*)outp)[o] = (_Float16)v;
      else                    ((float*)outp)[o] += v;
    }
  }
}

// -------------------------------------------------------------------------
// fused pair kernel: Wij = ssp(rbf@fW1+b1)@fW2+b2, *rcut, gather h[idx_j],
// scatter-add into agg[idx_i].  One wave handles 16 pairs (one WMMA M-tile).
// -------------------------------------------------------------------------
__global__ void __launch_bounds__(128) pair_filter_agg_kernel(
    const _Float16* __restrict__ rbf, const float* __restrict__ rcut,
    const _Float16* __restrict__ fW1t /*[128][32]*/, const float* __restrict__ fb1,
    const _Float16* __restrict__ fW2t /*[128][128]*/, const float* __restrict__ fb2,
    const float* __restrict__ h, const int* __restrict__ idx_i,
    const int* __restrict__ idx_j, float* __restrict__ agg) {
  __shared__ __align__(16) _Float16 smem[4][16][136];  // C->A layout shuffle
  const int wid = threadIdx.x >> 5;
  const int lane = threadIdx.x & 31;
  const int pairbase = (blockIdx.x * 4 + wid) * 16;
  const int col0 = lane & 15;
  const int rofs = (lane & 16) ? 8 : 0;
  const int aoff = (lane & 16) ? 8 : 0;
  const int boff = (lane & 16) ? 8 : 0;

  // GEMM1: [16x32] rbf-tile @ [32x128] fW1 (K=32, one WMMA per n-tile)
  {
    const _Float16* ap = rbf + (size_t)(pairbase + col0) * 32 + aoff;
    v16h a1 = ld2x8h(ap, ap + 16);
#pragma unroll
    for (int nt = 0; nt < 8; ++nt) {
      int col = nt * 16 + col0;
      float bv = fb1[col];
      v8f acc = {bv, bv, bv, bv, bv, bv, bv, bv};
      const _Float16* bp = fW1t + col * 32 + boff;
      v16h b = ld2x8h(bp, bp + 16);
      acc = __builtin_amdgcn_wmma_f32_16x16x32_f16(false, a1, false, b,
                                                   (short)0, acc, false, false);
#pragma unroll
      for (int r = 0; r < 8; ++r)
        smem[wid][r + rofs][col] = (_Float16)ssp_f(acc[r]);   // C layout -> row-major
    }
  }
  __syncthreads();

  // per-pair metadata (row of each C VGPR)
  float rc[8]; int jo[8]; int io[8];
#pragma unroll
  for (int r = 0; r < 8; ++r) {
    int pr = pairbase + r + rofs;
    rc[r] = rcut[pr];
    jo[r] = idx_j[pr] * NB;
    io[r] = idx_i[pr] * NB;
  }

  // A fragments for GEMM2 from LDS (row-major -> ISA A layout)
  v16h a2[4];
#pragma unroll
  for (int ks = 0; ks < 4; ++ks) {
    const _Float16* p = &smem[wid][col0][ks * 32 + aoff];
    a2[ks] = ld2x8h(p, p + 16);
  }

  // GEMM2 + cutoff + gather-multiply + atomic scatter
#pragma unroll
  for (int nt = 0; nt < 8; ++nt) {
    int col = nt * 16 + col0;
    float bv = fb2[col];
    v8f acc = {bv, bv, bv, bv, bv, bv, bv, bv};
#pragma unroll
    for (int ks = 0; ks < 4; ++ks) {
      const _Float16* bp = fW2t + col * 128 + ks * 32 + boff;
      v16h b = ld2x8h(bp, bp + 16);
      acc = __builtin_amdgcn_wmma_f32_16x16x32_f16(false, a2[ks], false, b,
                                                   (short)0, acc, false, false);
    }
#pragma unroll
    for (int r = 0; r < 8; ++r) {
      float w = acc[r] * rc[r];
      float v = h[jo[r] + col] * w;                 // coalesced 64B gather per half-wave
      unsafeAtomicAdd(&agg[io[r] + col], v);        // hw global_atomic_add_f32
    }
  }
}

// -------------------------------------------------------------------------
// host launcher
// -------------------------------------------------------------------------
extern "C" void kernel_launch(void* const* d_in, const int* in_sizes, int n_in,
                              void* d_out, int out_size, void* d_ws, size_t ws_size,
                              hipStream_t stream) {
  (void)in_sizes; (void)n_in; (void)out_size; (void)ws_size;
  const int*   Z     = (const int*)  d_in[0];
  const float* R     = (const float*)d_in[1];
  const int*   idx_i = (const int*)  d_in[2];
  const int*   idx_j = (const int*)  d_in[3];
  const float* emb   = (const float*)d_in[4];
  const float* in2fW = (const float*)d_in[5];
  const float* in2fb = (const float*)d_in[6];
  const float* fW1   = (const float*)d_in[7];
  const float* fb1   = (const float*)d_in[8];
  const float* fW2   = (const float*)d_in[9];
  const float* fb2   = (const float*)d_in[10];
  const float* oW1   = (const float*)d_in[11];
  const float* ob1   = (const float*)d_in[12];
  const float* oW2   = (const float*)d_in[13];
  const float* ob2   = (const float*)d_in[14];
  float* x = (float*)d_out;   // residual stream lives in d_out

  // workspace layout
  char* ws = (char*)d_ws;
  _Float16* rbf  = (_Float16*)ws;                                   // NP*32 f16
  float*    rcut = (float*)(ws + (size_t)NP * 64);                  // NP f32
  float*    h    = (float*)(ws + (size_t)NP * 64 + (size_t)NP * 4); // NATOMS*NB f32
  float*    agg  = h + (size_t)NATOMS * NB;                         // NATOMS*NB f32
  _Float16* u    = (_Float16*)(agg + (size_t)NATOMS * NB);          // NATOMS*NB f16
  _Float16* wbase = (_Float16*)((char*)u + (size_t)NATOMS * NB * 2);
  const size_t WPL = 16384 + 4096 + 16384 + 16384 + 16384;          // halves per layer

  // one-time (per call) precompute
  pair_geom_kernel<<<NP / 256, 256, 0, stream>>>(R, idx_i, idx_j, rbf, rcut);
  embed_kernel<<<(NATOMS * NB) / 256, 256, 0, stream>>>(Z, emb, x);
  for (int l = 0; l < 3; ++l) {
    _Float16* wl = wbase + (size_t)l * WPL;
    convw128_kernel<<<64, 256, 0, stream>>>(in2fW + l * 16384, wl);
    convw1_kernel  <<<16, 256, 0, stream>>>(fW1 + l * (NRBF * NB), wl + 16384);
    convw128_kernel<<<64, 256, 0, stream>>>(fW2 + l * 16384, wl + 16384 + 4096);
    convw128_kernel<<<64, 256, 0, stream>>>(oW1 + l * 16384, wl + 16384 + 4096 + 16384);
    convw128_kernel<<<64, 256, 0, stream>>>(oW2 + l * 16384, wl + 16384 + 4096 + 16384 + 16384);
  }

  for (int l = 0; l < 3; ++l) {
    _Float16* wl   = wbase + (size_t)l * WPL;
    _Float16* inWt = wl;
    _Float16* w1t  = wl + 16384;
    _Float16* w2t  = w1t + 4096;
    _Float16* o1t  = w2t + 16384;
    _Float16* o2t  = o1t + 16384;

    // h = x @ in2f_W + b
    wmma_gemm_k128<false, false, 0><<<128, 256, 0, stream>>>(x, inWt, in2fb + l * NB, h, NATOMS);
    // agg = 0
    hipMemsetAsync(agg, 0, (size_t)NATOMS * NB * 4, stream);
    // fused filter-network + gather + segment_sum
    pair_filter_agg_kernel<<<NP / 64, 128, 0, stream>>>(
        rbf, rcut, w1t, fb1 + l * NB, w2t, fb2 + l * NB, h, idx_i, idx_j, agg);
    // u = ssp(agg @ oW1 + b)   (f16)
    wmma_gemm_k128<false, true, 1><<<128, 256, 0, stream>>>(agg, o1t, ob1 + l * NB, u, NATOMS);
    // x += u @ oW2 + b
    wmma_gemm_k128<true, false, 2><<<128, 256, 0, stream>>>(u, o2t, ob2 + l * NB, x, NATOMS);
  }
}
